// SparseAttentionn_5712306504188
// MI455X (gfx1250) — matrix operand order repaired and checked by reading, not compiled, not run on hardware
//
#include <hip/hip_runtime.h>

// ---------------------------------------------------------------------------
// Dense attention (the reference's strided permute cancels out exactly):
//   O[b] = softmax(Q[b] K[b]^T * 0.125) V[b],  b in 0..3, T=4096, D=1024.
// Flash-attention single pass, bf16 WMMA (v_wmma_f32_16x16x32_bf16) with
// fp32 accumulation and fp32 online softmax. Wave32 / WGP tuned.
// ---------------------------------------------------------------------------

typedef __bf16         v16bf __attribute__((ext_vector_type(16)));
typedef float          v8f   __attribute__((ext_vector_type(8)));
typedef unsigned int   v8u   __attribute__((ext_vector_type(8)));
typedef unsigned short ushort_t;

constexpr int   T          = 4096;
constexpr int   D          = 1024;
constexpr int   NB         = 4;
constexpr int   BM         = 64;    // query rows per workgroup
constexpr int   BN         = 64;    // keys per tile
constexpr int   NTHREADS   = 256;   // 8 waves (wave32)
constexpr float ATTN_SCALE = 0.125f; // 1/sqrt(1024/16)

// LDS element strides, padded so WMMA fragment reads spread across banks.
constexpr int QS = D + 8;    // 1032 ushorts -> 516 dwords, 516 % 64 = 4
constexpr int KS = D + 8;
constexpr int VS = BN + 8;   // 72 ushorts -> 36 dwords
constexpr int PS = BN + 8;
constexpr int SS = BN + 4;   // 68 floats

constexpr size_t Q_OFF    = 0;
constexpr size_t Q_BYTES  = (size_t)BM * QS * 2;                 // 132096
constexpr size_t KV_OFF   = Q_OFF + Q_BYTES;
constexpr size_t K_BYTES  = (size_t)BN * KS * 2;                 // 132096
constexpr size_t V_BYTES  = (size_t)D * VS * 2;                  // 147456
constexpr size_t KV_BYTES = (K_BYTES > V_BYTES) ? K_BYTES : V_BYTES;
constexpr size_t S_OFF    = KV_OFF + KV_BYTES;
constexpr size_t S_BYTES  = (size_t)BM * SS * 4;                 // 17408
constexpr size_t P_OFF    = S_OFF + S_BYTES;
constexpr size_t P_BYTES  = (size_t)BM * PS * 2;                 // 9216
constexpr size_t M_OFF    = P_OFF + P_BYTES;
constexpr size_t L_OFF    = M_OFF + (size_t)BM * 4;
constexpr size_t SC_OFF   = L_OFF + (size_t)BM * 4;
constexpr size_t LDS_BYTES = SC_OFF + (size_t)BM * 4;            // ~307 KB < 320 KB

static __device__ __forceinline__ ushort_t f2bf(float f) {
  unsigned u = __builtin_bit_cast(unsigned, f);
  u += 0x7fffu + ((u >> 16) & 1u);   // round-to-nearest-even
  return (ushort_t)(u >> 16);
}

// Build a v16bf WMMA operand from two contiguous 16B LDS chunks.
static __device__ __forceinline__ v16bf ldsfrag(const ushort_t* p0, const ushort_t* p1) {
  uint4 a = *(const uint4*)p0;
  uint4 b = *(const uint4*)p1;
  v8u v = {a.x, a.y, a.z, a.w, b.x, b.y, b.z, b.w};
  return __builtin_bit_cast(v16bf, v);
}

template <int XM>
static __device__ __forceinline__ float swz_xor(float x) {
  int i = __builtin_bit_cast(int, x);
  i = __builtin_amdgcn_ds_swizzle(i, 0x1f | (XM << 10)); // group-of-32 xor
  return __builtin_bit_cast(float, i);
}

#define WMMA_BF16(A, B, C) \
  __builtin_amdgcn_wmma_f32_16x16x32_bf16(false, (A), false, (B), (short)0, (C), false, false)

__global__ void __launch_bounds__(NTHREADS)
flash_attn_bf16_kernel(const float* __restrict__ q, const float* __restrict__ k,
                       const float* __restrict__ v, float* __restrict__ out) {
  extern __shared__ char smem[];
  ushort_t* Qb  = (ushort_t*)(smem + Q_OFF);   // [BM][QS] bf16, Q block
  ushort_t* KVb = (ushort_t*)(smem + KV_OFF);  // K tile [BN][KS]  OR  V^T tile [D][VS]
  float*    Sf  = (float*)(smem + S_OFF);      // [BM][SS] raw scores (fp32)
  ushort_t* Pb  = (ushort_t*)(smem + P_OFF);   // [BM][PS] probabilities (bf16)
  float*    mL  = (float*)(smem + M_OFF);      // running row max
  float*    lL  = (float*)(smem + L_OFF);      // running row sum
  float*    scL = (float*)(smem + SC_OFF);     // per-iteration rescale

  const int tid  = threadIdx.x;
  const int w    = tid >> 5;        // wave id 0..7
  const int lane = tid & 31;
  const int hl   = lane >> 4;       // half-wave select
  const int ln   = lane & 15;

  const int qb    = blockIdx.x;     // query block
  const int batch = blockIdx.y;

  const float* Qg = q + ((size_t)batch * T + (size_t)qb * BM) * D;
  const float* Kg = k + (size_t)batch * T * D;
  const float* Vg = v + (size_t)batch * T * D;
  float*       Og = out + ((size_t)batch * T + (size_t)qb * BM) * D;

  if (tid < BM) { mL[tid] = -1e30f; lL[tid] = 0.0f; }

  // ---- stage Q block fp32 -> bf16 (once) ----
  for (int i = tid; i < BM * D / 4; i += NTHREADS) {
    int r = i >> 8;                 // D/4 == 256
    int c4 = (i & 255) << 2;
    float4 f = ((const float4*)Qg)[i];
    ushort4 u;
    u.x = f2bf(f.x); u.y = f2bf(f.y); u.z = f2bf(f.z); u.w = f2bf(f.w);
    *(ushort4*)(Qb + r * QS + c4) = u;
  }

  const v8f vzero = {0.f, 0.f, 0.f, 0.f, 0.f, 0.f, 0.f, 0.f};
  v8f acc[4][8];                    // O accumulator: 4 row-blocks x 8 col-subtiles
  #pragma unroll
  for (int a = 0; a < 4; ++a)
    #pragma unroll
    for (int b = 0; b < 8; ++b) acc[a][b] = vzero;

  __syncthreads();

  const int rb  = w >> 1;           // S-phase row-block of this wave
  const int cb0 = (w & 1) * 2;      // S-phase first col-block of this wave

  for (int kt = 0; kt < T / BN; ++kt) {
    // ---- stage K tile fp32 -> bf16 ----
    const float* Kt = Kg + (size_t)kt * BN * D;
    for (int i = tid; i < BN * D / 4; i += NTHREADS) {
      int r = i >> 8;
      int c4 = (i & 255) << 2;
      float4 f = ((const float4*)Kt)[i];
      ushort4 u;
      u.x = f2bf(f.x); u.y = f2bf(f.y); u.z = f2bf(f.z); u.w = f2bf(f.w);
      *(ushort4*)(KVb + r * KS + c4) = u;
    }
    __syncthreads();

    // ---- S = Q K^T : two 16x16 tiles per wave, contract over D ----
    v8f s0 = vzero, s1 = vzero;
    const ushort_t* qrow = Qb + (rb * 16 + ln) * QS + hl * 8;
    const ushort_t* k0   = KVb + ((cb0 + 0) * 16 + ln) * KS + hl * 8;
    const ushort_t* k1   = KVb + ((cb0 + 1) * 16 + ln) * KS + hl * 8;
    #pragma unroll 4
    for (int c = 0; c < D / 32; ++c) {
      v16bf A  = ldsfrag(qrow + c * 32, qrow + c * 32 + 16); // K {0..7,16..23}/{8..15,24..31}
      v16bf B0 = ldsfrag(k0 + c * 32, k0 + c * 32 + 16);     // K {0..7,16..23}/{8..15,24..31}
      v16bf B1 = ldsfrag(k1 + c * 32, k1 + c * 32 + 16);
      s0 = WMMA_BF16(A, B0, s0);
      s1 = WMMA_BF16(A, B1, s1);
    }
    {
      int rbase = rb * 16 + hl * 8;
      #pragma unroll
      for (int p = 0; p < 8; ++p) {
        Sf[(rbase + p) * SS + (cb0 + 0) * 16 + ln] = s0[p] * ATTN_SCALE;
        Sf[(rbase + p) * SS + (cb0 + 1) * 16 + ln] = s1[p] * ATTN_SCALE;
      }
    }
    __syncthreads();

    // ---- online softmax for the 64 new columns (4 lanes per row) ----
    {
      int row = tid >> 2;
      int g   = tid & 3;
      const float* srow = Sf + row * SS + g * 16;
      float sv[16];
      float mx = -1e30f;
      #pragma unroll
      for (int j = 0; j < 16; ++j) { sv[j] = srow[j]; mx = fmaxf(mx, sv[j]); }
      mx = fmaxf(mx, swz_xor<1>(mx));
      mx = fmaxf(mx, swz_xor<2>(mx));
      float mold = mL[row];
      float mnew = fmaxf(mold, mx);
      float rescale = __expf(mold - mnew);
      float ls = 0.f;
      ushort_t* prow = Pb + row * PS + g * 16;
      #pragma unroll
      for (int j = 0; j < 16; ++j) {
        float pj = __expf(sv[j] - mnew);
        ls += pj;
        prow[j] = f2bf(pj);
      }
      ls += swz_xor<1>(ls);
      ls += swz_xor<2>(ls);
      if (g == 0) {
        mL[row] = mnew;
        lL[row] = lL[row] * rescale + ls;
        scL[row] = rescale;
      }
    }

    // ---- stage V tile transposed into the (now free) K buffer: [d][key] ----
    const float* Vt = Vg + (size_t)kt * BN * D;
    for (int i = tid; i < BN * D / 4; i += NTHREADS) {
      int r  = i & (BN - 1);        // key row (lane-fast -> conflict-friendly DS writes)
      int db = i >> 6;              // d / 4
      float4 f = ((const float4*)Vt)[(size_t)r * (D / 4) + db];
      int d4 = db << 2;
      KVb[(d4 + 0) * VS + r] = f2bf(f.x);
      KVb[(d4 + 1) * VS + r] = f2bf(f.y);
      KVb[(d4 + 2) * VS + r] = f2bf(f.z);
      KVb[(d4 + 3) * VS + r] = f2bf(f.w);
    }
    __syncthreads();

    // prefetch next K tile into the cache hierarchy while we do P*V
    if (kt + 1 < T / BN) {
      const char* nk = (const char*)(Kg + (size_t)(kt + 1) * BN * D);
      #pragma unroll
      for (int i = 0; i < 8; ++i)
        __builtin_prefetch(nk + ((size_t)tid * 8 + i) * 128, 0, 3);
    }

    // ---- rescale running O, then O += P * V (wave owns 128 d-columns) ----
    #pragma unroll
    for (int r4 = 0; r4 < 4; ++r4) {
      float rs[8];
      #pragma unroll
      for (int p = 0; p < 8; ++p) rs[p] = scL[r4 * 16 + hl * 8 + p];
      #pragma unroll
      for (int s2 = 0; s2 < 8; ++s2)
        #pragma unroll
        for (int p = 0; p < 8; ++p) acc[r4][s2][p] *= rs[p];
    }
    v16bf Af[4][2];
    #pragma unroll
    for (int r4 = 0; r4 < 4; ++r4)
      #pragma unroll
      for (int kc = 0; kc < 2; ++kc) {
        const ushort_t* pr = Pb + (r4 * 16 + ln) * PS + kc * 32 + hl * 8;
        Af[r4][kc] = ldsfrag(pr, pr + 16);
      }
    #pragma unroll
    for (int s2 = 0; s2 < 8; ++s2) {
      const int dcol = w * 128 + s2 * 16 + ln;
      #pragma unroll
      for (int kc = 0; kc < 2; ++kc) {
        const ushort_t* vr = KVb + dcol * VS + kc * 32 + hl * 8;
        v16bf B = ldsfrag(vr, vr + 16);
        #pragma unroll
        for (int r4 = 0; r4 < 4; ++r4)
          acc[r4][s2] = WMMA_BF16(Af[r4][kc], B, acc[r4][s2]);
      }
    }
    __syncthreads();
  }

  // ---- normalize by row sums and store O (fp32) ----
  #pragma unroll
  for (int r4 = 0; r4 < 4; ++r4) {
    float inv[8];
    #pragma unroll
    for (int p = 0; p < 8; ++p) inv[p] = 1.0f / lL[r4 * 16 + hl * 8 + p];
    #pragma unroll
    for (int s2 = 0; s2 < 8; ++s2) {
      const int col = w * 128 + s2 * 16 + ln;
      #pragma unroll
      for (int p = 0; p < 8; ++p) {
        int row = r4 * 16 + hl * 8 + p;
        Og[(size_t)row * D + col] = acc[r4][s2][p] * inv[p];
      }
    }
  }
}

extern "C" void kernel_launch(void* const* d_in, const int* in_sizes, int n_in,
                              void* d_out, int out_size, void* d_ws, size_t ws_size,
                              hipStream_t stream) {
  (void)in_sizes; (void)n_in; (void)out_size; (void)d_ws; (void)ws_size;
  const float* q = (const float*)d_in[0];
  const float* k = (const float*)d_in[1];
  const float* v = (const float*)d_in[2];
  float* out = (float*)d_out;

  dim3 grid(T / BM, NB);   // 64 query blocks x 4 batches
  dim3 block(NTHREADS);
  flash_attn_bf16_kernel<<<grid, block, LDS_BYTES, stream>>>(q, k, v, out);
}
